// ExtractGraph_40389872451948
// MI455X (gfx1250) — matrix operand order repaired and checked by reading, not compiled, not run on hardware
//
#include <hip/hip_runtime.h>

// ---------------------------------------------------------------------------
// ExtractGraph for MI455X (gfx1250, wave32).
// 96x96 grid, diagonal-only edges, threshold + dropout gated, COO output
// padded to 4*9216 with sentinel 9216. Latency-bound: 5 small kernels.
// Inter-block scan of 144 block sums done with two v_wmma_f32_16x16x32_f16
// ops in a single wave (matrix-unit scan: S = X*U, Off = Lstrict*T).
// ---------------------------------------------------------------------------

#define GRID 96
#define NN   9216          // nodes = 96*96
#define CAND 36864         // 4 * NN candidate slots == MAX_EDGES
#define HWDIM 192
#define SENT 9216.0f

typedef __attribute__((ext_vector_type(16))) _Float16 v16h;
typedef __attribute__((ext_vector_type(8)))  float    v8f;

// ---- K1: 2x2 maxpool, d = pool + noise, per-block min/max of pool ---------
__global__ void k_pool(const float* __restrict__ dc,
                       const float* __restrict__ noise,
                       float* __restrict__ d,
                       float* __restrict__ mnp, float* __restrict__ mxp) {
    int g = blockIdx.x * 256 + threadIdx.x;          // 36 blocks * 256 = 9216
    int y = g / GRID, x = g - y * GRID;
    const float* r0 = dc + (2 * y) * HWDIM + 2 * x;
    float p = fmaxf(fmaxf(r0[0], r0[1]), fmaxf(r0[HWDIM], r0[HWDIM + 1]));
    d[g] = p + noise[g];

    __shared__ float smin[256], smax[256];
    smin[threadIdx.x] = p;
    smax[threadIdx.x] = p;
    __syncthreads();
    for (int s = 128; s > 0; s >>= 1) {
        if (threadIdx.x < s) {
            smin[threadIdx.x] = fminf(smin[threadIdx.x], smin[threadIdx.x + s]);
            smax[threadIdx.x] = fmaxf(smax[threadIdx.x], smax[threadIdx.x + s]);
        }
        __syncthreads();
    }
    if (threadIdx.x == 0) { mnp[blockIdx.x] = smin[0]; mxp[blockIdx.x] = smax[0]; }
}

// ---- K2: reduce 36 partials -> threshold = (max-min)/96 -------------------
__global__ void k_thresh(const float* __restrict__ mnp,
                         const float* __restrict__ mxp,
                         float* __restrict__ thr) {
    if (threadIdx.x == 0) {
        float mn = mnp[0], mx = mxp[0];
        for (int i = 1; i < 36; ++i) { mn = fminf(mn, mnp[i]); mx = fmaxf(mx, mxp[i]); }
        thr[0] = (mx - mn) / 96.0f;
    }
}

// ---- K3: edge flags per candidate slot + per-block counts + sentinel fill -
// slot t = 4*i + k; k in label-ascending neighbor order:
//   k=0:(-1,-1) j=i-97   k=1:(-1,+1) j=i-95   k=2:(+1,-1) j=i+95   k=3:(+1,+1) j=i+97
__global__ void k_flags(const float* __restrict__ d,
                        const unsigned char* __restrict__ mask,
                        const float* __restrict__ thr,
                        int* __restrict__ flags,
                        float* __restrict__ bsum,
                        float* __restrict__ out) {
    int t = blockIdx.x * 256 + threadIdx.x;          // 144 blocks * 256 = 36864
    // sentinel fill (out_size = 2*CAND); scatter kernel runs later on stream
    out[t] = SENT;
    out[CAND + t] = SENT;

    int i = t >> 2, k = t & 3;
    int y = i / GRID, x = i - y * GRID;
    int ny = y + ((k < 2) ? -1 : 1);
    int nx = x + ((k & 1) ? 1 : -1);

    __builtin_prefetch(mask + (size_t)i * NN + (unsigned)(i > 97 ? i - 97 : 0), 0, 0);

    int f = 0;
    if ((unsigned)ny < (unsigned)GRID && (unsigned)nx < (unsigned)GRID) {
        int j = ny * GRID + nx;
        float th = thr[0];
        if (fabsf(d[j] - d[i]) <= th)
            f = mask[(size_t)i * NN + (size_t)j] ? 1 : 0;
    }
    flags[t] = f;

    __shared__ int ss[256];
    ss[threadIdx.x] = f;
    __syncthreads();
    for (int s = 128; s > 0; s >>= 1) {
        if (threadIdx.x < s) ss[threadIdx.x] += ss[threadIdx.x + s];
        __syncthreads();
    }
    if (threadIdx.x == 0) bsum[blockIdx.x] = (float)ss[0];
}

// ---- K4: exclusive scan of 144 block sums in ONE wave via WMMA ------------
// Pack sums as X[16][16] (idx = r*16+c, zero-padded). Two matrix ops:
//   S   = X * U        (U[k][n] = k<=n, f16)   -> row-wise inclusive scans
//   Off = Lstrict * T  (T[k] = S[k][15])       -> cross-row exclusive offsets
// excl[idx] = S[r][c] + Off[r] - x[idx].  Counts <=256: exact in f16/f32.
__global__ void k_scan(const float* __restrict__ bsum,
                       float* __restrict__ boff) {
    int L    = threadIdx.x;      // wave32: 0..31
    int Mrow = L & 15;
    int half = L >> 4;

    // A-matrix layout (16-bit 16x32): lane element j<8 -> K = half*8 + j (<16);
    // elements j>=8 -> K >= 16 (zero padding region of X).
    v16h a = {};
    for (int j = 0; j < 8; ++j) {
        int K   = half * 8 + j;
        int idx = Mrow * 16 + K;
        a[j] = (_Float16)((idx < 144) ? bsum[idx] : 0.0f);
    }
    // B-matrix layout (32x16): lane element j<8 -> K = half*8 + j; j>=8 -> K >= 16, N = L&15.
    v16h b = {};
    for (int j = 0; j < 16; ++j)
        b[j] = (_Float16)(((j < 8) && (half * 8 + j <= (L & 15))) ? 1.0f : 0.0f);

    v8f c0 = {};
    v8f s = __builtin_amdgcn_wmma_f32_16x16x32_f16(false, a, false, b,
                                                   (short)0, c0, false, false);
    // C/D layout: lane L, VGPR v -> D[M = v + 8*half][N = L&15]
    float sarr[8];
    for (int v = 0; v < 8; ++v) sarr[v] = s[v];

    // T[r] = S[r][15]: lane 15 holds r=0..7 (vgpr r), lane 31 holds r=8..15.
    v16h b2 = {};
    for (int j = 0; j < 16; ++j) {
        float tv = (j < 8) ? __shfl(sarr[j], 15, 32) : __shfl(sarr[j - 8], 31, 32);
        b2[j] = (_Float16)((half == 0) ? tv : 0.0f);   // B2[k][n] = T[k], k<16
    }
    v16h a2 = {};                                       // Lstrict[m][k] = k<m
    for (int j = 0; j < 16; ++j) {
        int K = half * 16 + j;
        a2[j] = (_Float16)((K < Mrow) ? 1.0f : 0.0f);
    }
    v8f off = __builtin_amdgcn_wmma_f32_16x16x32_f16(false, a2, false, b2,
                                                     (short)0, c0, false, false);

    for (int v = 0; v < 8; ++v) {
        int r = v + 8 * half;
        int idx = r * 16 + (L & 15);
        if (idx < 144) {
            float incl = sarr[v] + off[v];
            boff[idx] = incl - bsum[idx];              // exclusive offset
        }
    }
}

// ---- K5: ballot-based intra-block compaction + scatter --------------------
__global__ void k_scatter(const int* __restrict__ flags,
                          const float* __restrict__ boff,
                          float* __restrict__ out) {
    int t = blockIdx.x * 256 + threadIdx.x;
    int f = flags[t];

    unsigned int bal = __builtin_amdgcn_ballot_w32(f != 0);   // wave32 mask
    int lane = threadIdx.x & 31;
    int wid  = threadIdx.x >> 5;                               // 8 waves/block
    int wpre = __popc(bal & ((1u << lane) - 1u));
    int wcnt = __popc(bal);

    __shared__ int wsum[8];
    if (lane == 0) wsum[wid] = wcnt;
    __syncthreads();
    int woff = 0;
    for (int w = 0; w < wid; ++w) woff += wsum[w];

    if (f) {
        int i = t >> 2, k = t & 3;
        int y = i / GRID, x = i - y * GRID;
        int ny = y + ((k < 2) ? -1 : 1);
        int nx = x + ((k & 1) ? 1 : -1);
        int j = ny * GRID + nx;
        int pos = (int)boff[blockIdx.x] + woff + wpre;
        out[pos]        = (float)i;    // rows
        out[CAND + pos] = (float)j;    // cols
    }
}

// ---------------------------------------------------------------------------
extern "C" void kernel_launch(void* const* d_in, const int* in_sizes, int n_in,
                              void* d_out, int out_size, void* d_ws, size_t ws_size,
                              hipStream_t stream) {
    const float*         dc    = (const float*)d_in[0];          // [1,192,192]
    const float*         noise = (const float*)d_in[1];          // [1,96,96]
    const unsigned char* mask  = (const unsigned char*)d_in[2];  // [9216,9216] bool
    // d_in[3] = R_scale (unused by reference math)
    (void)in_sizes; (void)n_in; (void)out_size; (void)ws_size;

    float* ws    = (float*)d_ws;
    float* d     = ws;                 // 9216 floats: pooled + noise
    float* mnp   = d + NN;             // 36 (padded to 64)
    float* mxp   = mnp + 64;           // 36 (padded to 64)
    float* thr   = mxp + 64;           // 1  (padded to 16)
    float* bsum  = thr + 16;           // 144 block edge counts (f32)
    float* boff  = bsum + 160;         // 144 exclusive block offsets
    int*   flags = (int*)(boff + 160); // 36864 flags
    float* out   = (float*)d_out;      // [2, 36864]

    k_pool   <<< 36, 256, 0, stream>>>(dc, noise, d, mnp, mxp);
    k_thresh <<<  1,  32, 0, stream>>>(mnp, mxp, thr);
    k_flags  <<<144, 256, 0, stream>>>(d, mask, thr, flags, bsum, out);
    k_scan   <<<  1,  32, 0, stream>>>(bsum, boff);
    k_scatter<<<144, 256, 0, stream>>>(flags, boff, out);
}
